// AttentionHelper_34660386079089
// MI455X (gfx1250) — compile-verified, matrix operand order checked by reading, not run on hardware
//
#include <hip/hip_runtime.h>
#include <hip/hip_bf16.h>

// ---------------------------------------------------------------------------
// Masked attention for MI455X (gfx1250, wave32, WMMA).
//
// Roofline: must write the full 268 MB attention matrix -> memory-bound
// (~14 us at 23.3 TB/s). Compute (34 GF) has huge headroom, so we use a
// two-term f16 split (hi+lo) with v_wmma_f32_16x16x32_f16 (3 WMMA per
// 16x16x32 step) to recover ~fp32 precision at ~3x the minimal WMMA count.
// Inputs (50 MB) are L2-resident (192 MB L2), so per-WG K/V rereads are cheap.
// ---------------------------------------------------------------------------

typedef __attribute__((ext_vector_type(16))) _Float16 v16h;
typedef __attribute__((ext_vector_type(8)))  float    v8f;

#define B_  16
#define C_  128
#define L_  2048
#define LP  2052      // padded LDS row length (floats): +4 kills bank conflicts

__device__ __forceinline__ void split_f32(float x, _Float16& hi, _Float16& lo) {
    hi = (_Float16)x;
    lo = (_Float16)(x - (float)hi);
}

__global__ __launch_bounds__(256)
void attn_qk_softmax_pv(const float* __restrict__ Qp,
                        const float* __restrict__ Kp,
                        const float* __restrict__ Vp,
                        const float* __restrict__ Mp,
                        float* __restrict__ outp,
                        float* __restrict__ attnT)
{
    extern __shared__ float smem[];
    float* P    = smem;              // [16][LP] logits -> probabilities
    float* scr1 = smem + 16 * LP;    // [16][16] row-max partials
    float* scr2 = scr1 + 256;        // [16][16] row-sum partials

    const int tid  = threadIdx.x;
    const int lane = tid & 31;       // wave32
    const int wave = tid >> 5;       // 8 waves
    const int half = lane >> 4;      // 0 / 1 (lane half, per WMMA layout)
    const int l16  = lane & 15;

    const int b  = blockIdx.x >> 7;          // batch
    const int q0 = (blockIdx.x & 127) << 4;  // 16-query tile origin

    const float scale = 0.08838834764831845f; // 1/sqrt(C)

    // =====================================================================
    // Stage A prologue: Q A-fragments (M=q, Kdim=c), resident in registers.
    // A layout (16-bit, 16x32): lanes<16 -> M=lane, elems 0..7 = K c0..c0+7,
    // elems 8..15 = K c0+16..23; lanes>=16 -> K offset +8.
    // =====================================================================
    v16h ahi[4], alo[4];
    {
        const int m = l16;
        #pragma unroll
        for (int c4 = 0; c4 < 4; ++c4) {
            const int cb = c4 * 32 + half * 8;
            #pragma unroll
            for (int e = 0; e < 8; ++e) {
                float x0 = Qp[(size_t)(b * C_ + cb + e)      * L_ + q0 + m];
                float x1 = Qp[(size_t)(b * C_ + cb + 16 + e) * L_ + q0 + m];
                _Float16 h, l;
                split_f32(x0, h, l); ahi[c4][e]     = h; alo[c4][e]     = l;
                split_f32(x1, h, l); ahi[c4][8 + e] = h; alo[c4][8 + e] = l;
            }
        }
    }

    // =====================================================================
    // Stage A: logits. Each wave does 16 k-tiles, in adjacent pairs so the
    // wave consumes full 128B lines of K.
    // B fragment (Kdim=c, N=k): lanes pick column k = k0 + l16; lane pairs
    // (l, l+16) share k, covering c-rows c0..c0+7 / +8..15 etc.
    // =====================================================================
    for (int p = 0; p < 8; ++p) {
        #pragma unroll
        for (int sub = 0; sub < 2; ++sub) {
            const int kt   = ((p * 8 + wave) << 1) | sub;   // 0..127, disjoint
            const int k0   = kt << 4;
            const int kcol = k0 + l16;
            const float lm = __logf(Mp[b * L_ + kcol] + 1e-6f); // additive log-mask

            v8f acc = {};
            #pragma unroll
            for (int c4 = 0; c4 < 4; ++c4) {
                const int cb = c4 * 32 + half * 8;
                v16h bhi, blo;
                #pragma unroll
                for (int e = 0; e < 8; ++e) {
                    float x0 = Kp[(size_t)(b * C_ + cb + e)      * L_ + kcol];
                    float x1 = Kp[(size_t)(b * C_ + cb + 16 + e) * L_ + kcol];
                    _Float16 h, l;
                    split_f32(x0, h, l); bhi[e]     = h; blo[e]     = l;
                    split_f32(x1, h, l); bhi[8 + e] = h; blo[8 + e] = l;
                }
                // two-term split product: hi*hi + hi*lo + lo*hi (~fp32)
                acc = __builtin_amdgcn_wmma_f32_16x16x32_f16(false, ahi[c4], false, bhi, (short)0, acc, false, false);
                acc = __builtin_amdgcn_wmma_f32_16x16x32_f16(false, ahi[c4], false, blo, (short)0, acc, false, false);
                acc = __builtin_amdgcn_wmma_f32_16x16x32_f16(false, alo[c4], false, bhi, (short)0, acc, false, false);
            }
            // D layout: VGPR r -> row m = r + 8*half, col n = l16. Store
            // scaled+log-masked logits; conflict-free (LP padding).
            #pragma unroll
            for (int r = 0; r < 8; ++r) {
                const int m = r + half * 8;
                P[m * LP + k0 + l16] = acc[r] * scale + lm;
            }
        }
    }
    __syncthreads();

    // =====================================================================
    // Softmax over each 2048-wide row; 16 threads per row, k = s + 16j
    // (bank-conflict-free with LP=2052). Then normalize, apply
    // multiplicative mask, write attn^T coalesced (lanes 0..15 = contig q).
    // =====================================================================
    {
        const int q = tid & 15;
        const int s = tid >> 4;
        float* row = P + q * LP;

        float mx = -3.402823466e38f;
        #pragma unroll 8
        for (int j = 0; j < 128; ++j) mx = fmaxf(mx, row[s + 16 * j]);
        scr1[q * 16 + s] = mx;
        __syncthreads();

        float rmx = scr1[q * 16];
        #pragma unroll
        for (int i = 1; i < 16; ++i) rmx = fmaxf(rmx, scr1[q * 16 + i]);

        float sum = 0.f;
        #pragma unroll 8
        for (int j = 0; j < 128; ++j) {
            const int k = s + 16 * j;
            float e = __expf(row[k] - rmx);
            row[k] = e;
            sum += e;
        }
        scr2[q * 16 + s] = sum;
        __syncthreads();

        float rs = 0.f;
        #pragma unroll
        for (int i = 0; i < 16; ++i) rs += scr2[q * 16 + i];
        const float inv = 1.0f / rs;

        #pragma unroll 4
        for (int j = 0; j < 128; ++j) {
            const int k = s + 16 * j;
            float v = row[k] * inv * Mp[b * L_ + k];   // post-softmax mask
            row[k] = v;                                 // keep for PV stage
            attnT[((size_t)(b * L_ + k)) * L_ + q0 + q] = v;
        }
    }
    __syncthreads();

    // =====================================================================
    // Stage B: out[c,q] = sum_k V[c,k] * P[q,k].  A = V (M=c, Kdim=k),
    // B = P^T (Kdim=k, N=q).  This orientation gives coalesced out stores
    // (lanes 0..15 write contiguous q) with no extra staging. Lane pairs
    // (l, l+16) share c, so each chunk reads full 128B lines of V.
    // =====================================================================
    {
        const int cb = wave * 16;
        const int c  = cb + l16;
        const float* vrow = Vp + (size_t)(b * C_ + c) * L_;
        const float* prow = P + l16 * LP;    // column q = l16 of B

        v8f acc = {};
        for (int ch = 0; ch < 64; ++ch) {
            const int k0 = ch * 32;
            const int kh = k0 + half * 8;
            v16h vhi, vlo, phi, plo;
            #pragma unroll
            for (int e = 0; e < 8; ++e) {
                _Float16 h, l;
                float x0 = vrow[kh + e];
                float x1 = vrow[kh + 16 + e];
                split_f32(x0, h, l); vhi[e]     = h; vlo[e]     = l;
                split_f32(x1, h, l); vhi[8 + e] = h; vlo[8 + e] = l;
                float p0 = prow[kh + e];
                float p1 = prow[kh + 16 + e];
                split_f32(p0, h, l); phi[e]     = h; plo[e]     = l;
                split_f32(p1, h, l); phi[8 + e] = h; plo[8 + e] = l;
            }
            acc = __builtin_amdgcn_wmma_f32_16x16x32_f16(false, vhi, false, phi, (short)0, acc, false, false);
            acc = __builtin_amdgcn_wmma_f32_16x16x32_f16(false, vhi, false, plo, (short)0, acc, false, false);
            acc = __builtin_amdgcn_wmma_f32_16x16x32_f16(false, vlo, false, phi, (short)0, acc, false, false);
        }
        // D: VGPR r -> c = cb + r + 8*half, q = q0 + l16 (coalesced store)
        #pragma unroll
        for (int r = 0; r < 8; ++r) {
            const int cc = cb + r + half * 8;
            outp[(size_t)(b * C_ + cc) * L_ + q0 + l16] = acc[r];
        }
    }
}

extern "C" void kernel_launch(void* const* d_in, const int* in_sizes, int n_in,
                              void* d_out, int out_size, void* d_ws, size_t ws_size,
                              hipStream_t stream) {
    const float* Qp = (const float*)d_in[0];
    const float* Kp = (const float*)d_in[1];
    const float* Vp = (const float*)d_in[2];
    const float* Mp = (const float*)d_in[3];

    float* outp  = (float*)d_out;                      // [B, C, L]
    float* attnT = outp + (size_t)B_ * C_ * L_;        // [B, L, L]

    const size_t shmem = (size_t)(16 * LP + 512) * sizeof(float); // ~130.3 KB

    // Allow >64KB dynamic LDS (WGP has 320 KB; this leaves 2 WGs/WGP).
    (void)hipFuncSetAttribute((const void*)attn_qk_softmax_pv,
                              hipFuncAttributeMaxDynamicSharedMemorySize,
                              (int)shmem);

    attn_qk_softmax_pv<<<dim3(B_ * (L_ / 16)), dim3(256), shmem, stream>>>(
        Qp, Kp, Vp, Mp, outp, attnT);
}
